// HybridStockModel_23021024707550
// MI455X (gfx1250) — matrix operand order repaired and checked by reading, not compiled, not run on hardware
//
#include <hip/hip_runtime.h>
#include <cstdint>
#include <cstddef>

typedef __attribute__((ext_vector_type(16))) _Float16 v16h;
typedef __attribute__((ext_vector_type(8)))  float    v8f;

__device__ inline int lane_id() { return threadIdx.x & 31; }

__device__ inline v8f wm(v16h a, v16h b, v8f c) {
  return __builtin_amdgcn_wmma_f32_16x16x32_f16(false, a, false, b, (short)0, c, false, false);
}

// A-matrix 16x32 f16 tile (row-major source, A pre-offset to tile origin).
// lane<16: row=lane, K=0..15; lane>=16: row=lane-16, K=16..31
__device__ inline v16h ld_a(const _Float16* A, int lda) {
  int lane = lane_id();
  const _Float16* p = A + (lane & 15) * lda + ((lane >> 4) << 4);
  v16h r;
#pragma unroll
  for (int i = 0; i < 8; ++i) r[i] = p[i];
#pragma unroll
  for (int i = 0; i < 8; ++i) r[8 + i] = p[8 + i];
  return r;
}

// B-matrix 32x16 f16 tile from TRANSPOSED storage: Bt[col][k] (row-major N x K).
// lane<16: col=lane, K=0..15 ; lane>=16: col=lane-16, K=16..31  (contiguous 16 halfs/lane)
__device__ inline v16h ld_bT(const _Float16* Bt, int ldb) {
  int lane = lane_id();
  const _Float16* p = Bt + (lane & 15) * ldb + ((lane >> 4) << 4);
  v16h r;
#pragma unroll
  for (int e = 0; e < 16; ++e) r[e] = p[e];
  return r;
}

// B tile from row-major K x N storage (strided; used only for tiny MHA V tiles)
__device__ inline v16h ld_b(const _Float16* B, int ldb) {
  int lane = lane_id();
  const _Float16* p = B + ((lane >> 4) << 4) * ldb + (lane & 15);
  v16h r;
#pragma unroll
  for (int e = 0; e < 16; ++e) r[e] = p[e * ldb];
  return r;
}

__device__ inline float wred_sum(float v) {
#pragma unroll
  for (int o = 16; o; o >>= 1) v += __shfl_xor(v, o, 32);
  return v;
}
__device__ inline float sigmoidf_(float x) { return 1.f / (1.f + expf(-x)); }

__device__ inline void ln64(float v0, float v1, const float* g, const float* b,
                            float& y0, float& y1) {
  int lane = lane_id();
  float mu = wred_sum(v0 + v1) * (1.f / 64.f);
  float d0 = v0 - mu, d1 = v1 - mu;
  float inv = rsqrtf(wred_sum(d0 * d0 + d1 * d1) * (1.f / 64.f) + 1e-5f);
  y0 = d0 * inv * g[lane] + b[lane];
  y1 = d1 * inv * g[lane + 32] + b[lane + 32];
}

// ---------------- generic WMMA GEMM: one wave per 16x16 tile ----------------
// B is always in transposed (N x K) storage -> contiguous b128 loads.
// act: 0 none, 1 gelu(exact), 2 retanh, 3 tanh.  c16T: store f16 mirror transposed.
__global__ void __launch_bounds__(128)
k_gemm(const _Float16* __restrict__ A, int a_ld,
       const _Float16* __restrict__ Bt, int b_ld,
       const float* __restrict__ bias, const float* __restrict__ bias2,
       float* __restrict__ C, int c_ld,
       _Float16* __restrict__ C16, int c16_ld, int c16T,
       int M, int Nc, int K, int act, int accum)
{
  int ntn = Nc >> 4;
  int tile = blockIdx.x * 4 + (threadIdx.x >> 5);
  if (tile >= (M >> 4) * ntn) return;
  int tm = tile / ntn, tn = tile - tm * ntn;
  int lane = lane_id();
  int l15 = lane & 15, rb = (lane >> 4) << 3;
  const _Float16* Ab = A + (size_t)(tm * 16) * a_ld;
  const _Float16* Bb = Bt + (size_t)(tn * 16) * b_ld;
  v8f c = {};
  for (int k0 = 0; k0 < K; k0 += 32) {
    v16h a = ld_a(Ab + k0, a_ld);
    v16h b = ld_bT(Bb + k0, b_ld);
    c = wm(a, b, c);
  }
  int col = tn * 16 + l15;
  float bv = (bias ? bias[col] : 0.f) + (bias2 ? bias2[col] : 0.f);
#pragma unroll
  for (int i = 0; i < 8; ++i) {
    int row = tm * 16 + rb + i;
    float v = c[i] + bv;
    if (act == 1) v = 0.5f * v * (1.f + erff(v * 0.70710678f));
    else if (act == 2) v = (v <= 0.f) ? 0.f : tanhf(v);
    else if (act == 3) v = tanhf(v);
    if (C) {
      size_t idx = (size_t)row * c_ld + col;
      if (accum) v += C[idx];
      C[idx] = v;
    }
    if (C16) {
      size_t idx = c16T ? ((size_t)col * c16_ld + row) : ((size_t)row * c16_ld + col);
      C16[idx] = (_Float16)v;
    }
  }
}

// ---------------- input layernorm + projection (K=12) ----------------
__global__ void k_inproj(const float* __restrict__ feat,
                         const float* __restrict__ lnw, const float* __restrict__ lnb,
                         const float* __restrict__ pw, const float* __restrict__ pb,
                         float* __restrict__ x32, _Float16* __restrict__ x16)
{
  int r = blockIdx.x * 256 + threadIdx.x; // 65536 rows
  float f[12];
  float mu = 0.f;
#pragma unroll
  for (int k = 0; k < 12; ++k) { f[k] = feat[(size_t)r * 12 + k]; mu += f[k]; }
  mu *= (1.f / 12.f);
  float var = 0.f;
#pragma unroll
  for (int k = 0; k < 12; ++k) { float d = f[k] - mu; var += d * d; }
  float inv = rsqrtf(var * (1.f / 12.f) + 1e-5f);
#pragma unroll
  for (int k = 0; k < 12; ++k) f[k] = (f[k] - mu) * inv * lnw[k] + lnb[k];
  for (int cc = 0; cc < 64; ++cc) {
    float a = pb[cc];
#pragma unroll
    for (int k = 0; k < 12; ++k) a += f[k] * pw[k * 64 + cc];
    x32[(size_t)r * 64 + cc] = a;
    x16[(size_t)r * 64 + cc] = (_Float16)a;
  }
}

// ---------------- fused persistent GRU: one wave = 16 rows, full 32 steps ----------------
// wihT/whhT are transposed (192 x 64) -> all B loads contiguous.
__global__ void __launch_bounds__(128)
k_gru(const _Float16* __restrict__ x16,
      const _Float16* __restrict__ wihT, const _Float16* __restrict__ whhT,
      const float* __restrict__ bih, const float* __restrict__ bhh,
      _Float16* __restrict__ zout, int reverse)
{
  __shared__ _Float16 hbuf[4][16 * 64];
  int wid = threadIdx.x >> 5;
  int lane = lane_id();
  int n0 = (blockIdx.x * 4 + wid) * 16;
  _Float16* hl = hbuf[wid];
  for (int i = lane; i < 16 * 64; i += 32) hl[i] = (_Float16)0.f;
  int l15 = lane & 15, rb = (lane >> 4) << 3;
  v8f hprev[4] = {{}, {}, {}, {}};
  for (int tt = 0; tt < 32; ++tt) {
    int t = reverse ? (31 - tt) : tt;
    const _Float16* xA = x16 + ((size_t)n0 * 32 + t) * 64;
    v16h ax0 = ld_a(xA, 2048);
    v16h ax1 = ld_a(xA + 32, 2048);
    v16h ah0 = ld_a(hl, 64);
    v16h ah1 = ld_a(hl + 32, 64);
#pragma unroll
    for (int jc = 0; jc < 4; ++jc) {
      int cr = jc * 16, cz = 64 + jc * 16, cn = 128 + jc * 16;
      const _Float16* wr = wihT + (size_t)cr * 64;
      const _Float16* wz = wihT + (size_t)cz * 64;
      const _Float16* wn = wihT + (size_t)cn * 64;
      const _Float16* ur = whhT + (size_t)cr * 64;
      const _Float16* uz = whhT + (size_t)cz * 64;
      const _Float16* un = whhT + (size_t)cn * 64;
      v8f gr = {}; gr = wm(ax0, ld_bT(wr, 64), gr); gr = wm(ax1, ld_bT(wr + 32, 64), gr);
      v8f gz = {}; gz = wm(ax0, ld_bT(wz, 64), gz); gz = wm(ax1, ld_bT(wz + 32, 64), gz);
      v8f gn = {}; gn = wm(ax0, ld_bT(wn, 64), gn); gn = wm(ax1, ld_bT(wn + 32, 64), gn);
      v8f hr = {}; hr = wm(ah0, ld_bT(ur, 64), hr); hr = wm(ah1, ld_bT(ur + 32, 64), hr);
      v8f hz = {}; hz = wm(ah0, ld_bT(uz, 64), hz); hz = wm(ah1, ld_bT(uz + 32, 64), hz);
      v8f hn = {}; hn = wm(ah0, ld_bT(un, 64), hn); hn = wm(ah1, ld_bT(un + 32, 64), hn);
      float bir = bih[cr + l15], biz = bih[cz + l15], bin_ = bih[cn + l15];
      float bhr = bhh[cr + l15], bhz = bhh[cz + l15], bhn = bhh[cn + l15];
      v8f hnew = {};
#pragma unroll
      for (int i = 0; i < 8; ++i) {
        float rg = sigmoidf_(gr[i] + bir + hr[i] + bhr);
        float zg = sigmoidf_(gz[i] + biz + hz[i] + bhz);
        float ng = tanhf(gn[i] + bin_ + rg * (hn[i] + bhn));
        hnew[i] = (1.f - zg) * ng + zg * hprev[jc][i];
      }
      hprev[jc] = hnew;
#pragma unroll
      for (int i = 0; i < 8; ++i) {
        int row = rb + i;
        _Float16 hv = (_Float16)hnew[i];
        hl[row * 64 + cr + l15] = hv;
        zout[(((size_t)(n0 + row)) * 32 + t) * 64 + cr + l15] = hv;
      }
    }
  }
}

// ---------------- gate combine + LN ----------------
__global__ void k_zg(const float* __restrict__ gateS,
                     const _Float16* __restrict__ zf, const _Float16* __restrict__ zb,
                     const float* __restrict__ x32,
                     const float* __restrict__ g, const float* __restrict__ b,
                     float* __restrict__ zg32, _Float16* __restrict__ zg16)
{
  int row = blockIdx.x * 8 + (threadIdx.x >> 5);
  int lane = lane_id();
  size_t base = (size_t)row * 64;
  float g0 = sigmoidf_(gateS[base + lane]);
  float g1 = sigmoidf_(gateS[base + 32 + lane]);
  float v0 = g0 * (float)zf[base + lane] + (1.f - g0) * (float)zb[base + lane] + x32[base + lane];
  float v1 = g1 * (float)zf[base + 32 + lane] + (1.f - g1) * (float)zb[base + 32 + lane] + x32[base + 32 + lane];
  float y0, y1;
  ln64(v0, v1, g, b, y0, y1);
  zg32[base + lane] = y0; zg32[base + 32 + lane] = y1;
  zg16[base + lane] = (_Float16)y0; zg16[base + 32 + lane] = (_Float16)y1;
}

// ---------------- generic row LayerNorm (64 wide) with optional residual ----------------
__global__ void k_rowln(const float* __restrict__ in, int in_ld,
                        const float* __restrict__ res, int res_ld,
                        const float* __restrict__ g, const float* __restrict__ b,
                        float* __restrict__ out32, _Float16* __restrict__ out16, int M)
{
  int row = blockIdx.x * 8 + (threadIdx.x >> 5);
  if (row >= M) return;
  int lane = lane_id();
  size_t base = (size_t)row * in_ld;
  float v0 = in[base + lane], v1 = in[base + 32 + lane];
  if (res) { size_t rbb = (size_t)row * res_ld; v0 += res[rbb + lane]; v1 += res[rbb + 32 + lane]; }
  float y0, y1;
  ln64(v0, v1, g, b, y0, y1);
  if (out32) { out32[(size_t)row * 64 + lane] = y0; out32[(size_t)row * 64 + 32 + lane] = y1; }
  if (out16) { out16[(size_t)row * 64 + lane] = (_Float16)y0; out16[(size_t)row * 64 + 32 + lane] = (_Float16)y1; }
}

// ---------------- MoE gating (softmax + argmax, first-max wins) ----------------
__global__ void k_moe_gate(const float* __restrict__ zg, const float* __restrict__ gw,
                           const float* __restrict__ gb, int* __restrict__ assign,
                           float* __restrict__ wsel)
{
  int r = blockIdx.x * 256 + threadIdx.x;
  float l[4];
#pragma unroll
  for (int e = 0; e < 4; ++e) {
    float a = gb[e];
    for (int k = 0; k < 64; ++k) a += zg[(size_t)r * 64 + k] * gw[k * 4 + e];
    l[e] = a;
  }
  int am = 0; float best = l[0];
#pragma unroll
  for (int e = 1; e < 4; ++e) if (l[e] > best) { best = l[e]; am = e; }
  float s = 0.f;
#pragma unroll
  for (int e = 0; e < 4; ++e) s += expf(l[e] - best);
  assign[r] = am;
  wsel[r] = 1.f / s; // exp(0)/sum
}

__global__ void k_moe_accum(const float* __restrict__ oute, const int* __restrict__ assign,
                            const float* __restrict__ wsel, float* __restrict__ zm, int e)
{
  size_t idx = (size_t)blockIdx.x * 256 + threadIdx.x; // 65536*64
  int r = (int)(idx >> 6);
  float v = (assign[r] == e) ? wsel[r] * oute[idx] : 0.f;
  if (e == 0) zm[idx] = v; else zm[idx] += v;
}

// ---------------- MHA: one wave per (node, head): QK^T -> softmax -> AV ----------------
__global__ void __launch_bounds__(128) k_mha(const _Float16* __restrict__ qkv,
                                             _Float16* __restrict__ za)
{
  __shared__ float    sc[4][32 * 32];
  __shared__ _Float16 at[4][32 * 32];
  int wid = threadIdx.x >> 5, lane = lane_id();
  int wave = blockIdx.x * 4 + wid;
  int n = wave >> 1, hh = wave & 1;
  const _Float16* base = qkv + (size_t)n * 32 * 192;
  const _Float16* q = base + hh * 32;
  const _Float16* km = base + 64 + hh * 32;
  const _Float16* vm = base + 128 + hh * 32;
  float* scl = sc[wid];
  _Float16* atl = at[wid];
  int l15 = lane & 15, rb = (lane >> 4) << 3;
  const float scale = 0.1767766952966369f; // 1/sqrt(32)
#pragma unroll
  for (int ti = 0; ti < 2; ++ti)
#pragma unroll
    for (int tj = 0; tj < 2; ++tj) {
      v8f c = {};
      c = wm(ld_a(q + ti * 16 * 192, 192), ld_bT(km + tj * 16 * 192, 192), c);
#pragma unroll
      for (int i = 0; i < 8; ++i)
        scl[(ti * 16 + rb + i) * 32 + tj * 16 + l15] = c[i] * scale;
    }
  __syncthreads();
  float mx = -1e30f;
#pragma unroll
  for (int j = 0; j < 32; ++j) mx = fmaxf(mx, scl[lane * 32 + j]);
  float sum = 0.f; float ex[32];
#pragma unroll
  for (int j = 0; j < 32; ++j) { ex[j] = expf(scl[lane * 32 + j] - mx); sum += ex[j]; }
  float inv = 1.f / sum;
#pragma unroll
  for (int j = 0; j < 32; ++j) atl[lane * 32 + j] = (_Float16)(ex[j] * inv);
  __syncthreads();
#pragma unroll
  for (int ti = 0; ti < 2; ++ti)
#pragma unroll
    for (int tj = 0; tj < 2; ++tj) {
      v8f c = {};
      c = wm(ld_a(atl + ti * 16 * 32, 32), ld_b(vm + tj * 16, 192), c);
#pragma unroll
      for (int i = 0; i < 8; ++i)
        za[((size_t)n * 32 + ti * 16 + rb + i) * 64 + hh * 32 + tj * 16 + l15] = (_Float16)c[i];
    }
}

// ---------------- GAT helpers ----------------
__global__ void k_gat_uv(const float* __restrict__ sp, const float* __restrict__ wu,
                         const float* __restrict__ wv, float* __restrict__ su,
                         float* __restrict__ sv)
{
  int n = blockIdx.x * 256 + threadIdx.x;
  if (n >= 2048) return;
#pragma unroll
  for (int h = 0; h < 8; ++h) {
    float a = 0.f, b = 0.f;
#pragma unroll
    for (int o = 0; o < 8; ++o) {
      float s = sp[(size_t)n * 64 + h * 8 + o];
      a += s * wu[h * 8 + o];
      b += s * wv[h * 8 + o];
    }
    su[h * 2048 + n] = a;
    sv[h * 2048 + n] = b;
  }
}

// online-softmax GAT row: one wave per (head, i); out += attn @ s
__global__ void __launch_bounds__(256) k_gat(const float* __restrict__ adj,
                                             const float* __restrict__ sp,
                                             const float* __restrict__ su,
                                             const float* __restrict__ sv,
                                             float* __restrict__ out)
{
  int wave = blockIdx.x * 8 + (threadIdx.x >> 5);
  int head = wave >> 11;
  int i = wave & 2047;
  int lane = lane_id();
  float fvi = sv[head * 2048 + i];
  float m = -1e30f, l = 0.f;
  float a0 = 0, a1 = 0, a2 = 0, a3 = 0, a4 = 0, a5 = 0, a6 = 0, a7 = 0;
  const float* arow = adj + (size_t)i * 2048;
  const float* suh = su + head * 2048;
  for (int j = lane; j < 2048; j += 32) {
    if (arow[j] > 0.f) {
      float e = suh[j] + fvi;
      e = (e > 0.f) ? e : 0.2f * e;
      float mn = fmaxf(m, e);
      float sc = expf(m - mn);
      float p = expf(e - mn);
      l = l * sc + p;
      const float* srow = sp + (size_t)j * 64 + head * 8;
      a0 = a0 * sc + p * srow[0]; a1 = a1 * sc + p * srow[1];
      a2 = a2 * sc + p * srow[2]; a3 = a3 * sc + p * srow[3];
      a4 = a4 * sc + p * srow[4]; a5 = a5 * sc + p * srow[5];
      a6 = a6 * sc + p * srow[6]; a7 = a7 * sc + p * srow[7];
      m = mn;
    }
  }
#pragma unroll
  for (int o = 16; o; o >>= 1) {
    float m2 = __shfl_xor(m, o, 32);
    float l2 = __shfl_xor(l, o, 32);
    float b0 = __shfl_xor(a0, o, 32), b1 = __shfl_xor(a1, o, 32);
    float b2 = __shfl_xor(a2, o, 32), b3 = __shfl_xor(a3, o, 32);
    float b4 = __shfl_xor(a4, o, 32), b5 = __shfl_xor(a5, o, 32);
    float b6 = __shfl_xor(a6, o, 32), b7 = __shfl_xor(a7, o, 32);
    float mn = fmaxf(m, m2);
    float s1 = expf(m - mn), s2 = expf(m2 - mn);
    l = l * s1 + l2 * s2;
    a0 = a0 * s1 + b0 * s2; a1 = a1 * s1 + b1 * s2;
    a2 = a2 * s1 + b2 * s2; a3 = a3 * s1 + b3 * s2;
    a4 = a4 * s1 + b4 * s2; a5 = a5 * s1 + b5 * s2;
    a6 = a6 * s1 + b6 * s2; a7 = a7 * s1 + b7 * s2;
    m = mn;
  }
  float inv = 1.f / l;
  if (lane < 8) {
    float val = (lane == 0) ? a0 : (lane == 1) ? a1 : (lane == 2) ? a2 : (lane == 3) ? a3
              : (lane == 4) ? a4 : (lane == 5) ? a5 : (lane == 6) ? a6 : a7;
    out[(size_t)i * 64 + head * 8 + lane] += val * inv;
  }
}

// ---------------- hypergraph helpers ----------------
__global__ void k_colsoftmax(const float* __restrict__ t1, float* __restrict__ Hs32,
                             _Float16* __restrict__ Hs16)
{
  __shared__ float red[256];
  int c = blockIdx.x, tid = threadIdx.x;
  float mx = -1e30f;
  for (int n = tid; n < 2048; n += 256) mx = fmaxf(mx, t1[n * 32 + c]);
  red[tid] = mx; __syncthreads();
  for (int s = 128; s; s >>= 1) { if (tid < s) red[tid] = fmaxf(red[tid], red[tid + s]); __syncthreads(); }
  mx = red[0]; __syncthreads();
  float sum = 0.f;
  for (int n = tid; n < 2048; n += 256) sum += expf(t1[n * 32 + c] - mx);
  red[tid] = sum; __syncthreads();
  for (int s = 128; s; s >>= 1) { if (tid < s) red[tid] += red[tid + s]; __syncthreads(); }
  float inv = 1.f / red[0];
  for (int n = tid; n < 2048; n += 256) {
    float v = expf(t1[n * 32 + c] - mx) * inv;
    Hs32[n * 32 + c] = v;
    Hs16[n * 32 + c] = (_Float16)v;
  }
}

__global__ void k_jsd(const float* __restrict__ Hs, float* __restrict__ jsd)
{
  int wave = blockIdx.x * 8 + (threadIdx.x >> 5); // 1024 pairs
  int a = wave >> 5, b = wave & 31;
  int lane = lane_id();
  float klp = 0.f, klq = 0.f;
  for (int n = lane; n < 2048; n += 32) {
    float p = Hs[n * 32 + a], q = Hs[n * 32 + b];
    float m = 0.5f * (p + q);
    const float eps = 1e-8f;
    klp += p * logf(fmaxf(p / (m + eps), eps));
    klq += q * logf(fmaxf(q / (m + eps), eps));
  }
  klp = wred_sum(klp); klq = wred_sum(klq);
  if (lane == 0) jsd[a * 32 + b] = fmaxf(0.5f * (klp + klq), 0.f);
}

__global__ void k_jsdw(const float* __restrict__ jsd, float* __restrict__ wvec)
{
  int a = threadIdx.x; // 32 threads = 1 wave
  float mu = 0.f;
  for (int b = 0; b < 32; ++b) mu += jsd[a * 32 + b];
  mu *= (1.f / 32.f);
  float mean = wred_sum(mu) * (1.f / 32.f);
  float d = mu - mean;
  float sd = sqrtf(wred_sum(d * d) * (1.f / 31.f)); // ddof=1
  float z = d / (sd + 1e-8f);
  float mx = z;
#pragma unroll
  for (int o = 16; o; o >>= 1) mx = fmaxf(mx, __shfl_xor(mx, o, 32));
  float e = expf(z - mx);
  float se = wred_sum(e);
  wvec[a] = e / se;
}

__global__ void k_hsw(const float* __restrict__ Hs, const float* __restrict__ wv,
                      _Float16* __restrict__ HsW)
{
  int i = blockIdx.x * 256 + threadIdx.x;
  if (i < 2048 * 32) HsW[i] = (_Float16)(Hs[i] * wv[i & 31]);
}

__global__ void k_elu(float* __restrict__ z)
{
  int i = blockIdx.x * 256 + threadIdx.x;
  if (i < 2048 * 64) { float v = z[i]; z[i] = (v > 0.f) ? v : (expf(v) - 1.f); }
}

// ---------------- semantic fusion ----------------
__global__ void k_pack(const float* __restrict__ h, const float* __restrict__ pos,
                       const float* __restrict__ neg, const float* __restrict__ hyp,
                       float* __restrict__ s32, _Float16* __restrict__ s16)
{
  int idx = blockIdx.x * 256 + threadIdx.x; // 2048*64
  if (idx >= 2048 * 64) return;
  int n = idx >> 6, c = idx & 63;
  float vs0 = h[idx], vs1 = pos[idx], vs2 = neg[idx], vs3 = hyp[idx];
  size_t base = (size_t)n * 4 * 64 + c;
  s32[base] = vs0;        s16[base] = (_Float16)vs0;
  s32[base + 64] = vs1;   s16[base + 64] = (_Float16)vs1;
  s32[base + 128] = vs2;  s16[base + 128] = (_Float16)vs2;
  s32[base + 192] = vs3;  s16[base + 192] = (_Float16)vs3;
}

__global__ void k_semlog(const _Float16* __restrict__ t2, const float* __restrict__ w2,
                         float* __restrict__ logits)
{
  int r = blockIdx.x * 8 + (threadIdx.x >> 5); // 8192 rows
  int lane = lane_id();
  float a = 0.f;
#pragma unroll
  for (int j = 0; j < 4; ++j) {
    int kk = lane + 32 * j;
    a += (float)t2[(size_t)r * 128 + kk] * w2[kk];
  }
  a = wred_sum(a);
  if (lane == 0) logits[r] = a;
}

__global__ void k_fuse(const float* __restrict__ logits, const float* __restrict__ s32,
                       float* __restrict__ fused)
{
  int n = blockIdx.x * 8 + (threadIdx.x >> 5);
  int lane = lane_id();
  float l0 = logits[n * 4 + 0], l1 = logits[n * 4 + 1];
  float l2 = logits[n * 4 + 2], l3 = logits[n * 4 + 3];
  float mx = fmaxf(fmaxf(l0, l1), fmaxf(l2, l3));
  float e0 = expf(l0 - mx), e1 = expf(l1 - mx), e2 = expf(l2 - mx), e3 = expf(l3 - mx);
  float inv = 1.f / (e0 + e1 + e2 + e3);
  float b0 = e0 * inv, b1 = e1 * inv, b2 = e2 * inv, b3 = e3 * inv;
  size_t base = (size_t)n * 4 * 64;
#pragma unroll
  for (int j = 0; j < 2; ++j) {
    int c = lane + 32 * j;
    fused[(size_t)n * 64 + c] = b0 * s32[base + c] + b1 * s32[base + 64 + c]
                              + b2 * s32[base + 128 + c] + b3 * s32[base + 192 + c];
  }
}

__global__ void k_colmean(const float* __restrict__ fused, float* __restrict__ colmu)
{
  __shared__ float red[256];
  int c = blockIdx.x, tid = threadIdx.x;
  float s = 0.f;
  for (int n = tid; n < 2048; n += 256) s += fused[(size_t)n * 64 + c];
  red[tid] = s; __syncthreads();
  for (int st = 128; st; st >>= 1) { if (tid < st) red[tid] += red[tid + st]; __syncthreads(); }
  if (tid == 0) colmu[c] = red[0] * (1.f / 2048.f);
}

__global__ void k_final(const float* __restrict__ fused, const float* __restrict__ colmu,
                        const float* __restrict__ pw, const float* __restrict__ pb,
                        float* __restrict__ out)
{
  int n = blockIdx.x * 8 + (threadIdx.x >> 5);
  int lane = lane_id();
  float v0 = fused[(size_t)n * 64 + lane] - colmu[lane];
  float v1 = fused[(size_t)n * 64 + 32 + lane] - colmu[32 + lane];
  float ss = wred_sum(v0 * v0 + v1 * v1);
  float dt = wred_sum(v0 * pw[lane] + v1 * pw[lane + 32]);
  if (lane == 0) out[n] = dt / sqrtf(1e-6f + ss) + pb[0];
}

// f32 (K x N) -> f16 transposed (N x K)
__global__ void k_f2hT(const float* __restrict__ s, _Float16* __restrict__ d, int K, int N)
{
  int i = blockIdx.x * 256 + threadIdx.x;
  if (i < K * N) {
    int k = i / N, n = i - k * N;
    d[(size_t)n * K + k] = (_Float16)s[i];
  }
}

// ======================================================================
extern "C" void kernel_launch(void* const* d_in, const int* in_sizes, int n_in,
                              void* d_out, int out_size, void* d_ws, size_t ws_size,
                              hipStream_t stream)
{
  (void)in_sizes; (void)n_in; (void)out_size; (void)ws_size;
  const float* features  = (const float*)d_in[0];
  const float* pos_adj   = (const float*)d_in[1];
  const float* neg_adj   = (const float*)d_in[2];
  const float* ln_in_w   = (const float*)d_in[3];
  const float* ln_in_b   = (const float*)d_in[4];
  const float* proj_w    = (const float*)d_in[5];
  const float* proj_b    = (const float*)d_in[6];
  const float* gru_wih_f = (const float*)d_in[7];
  const float* gru_whh_f = (const float*)d_in[8];
  const float* gru_bih_f = (const float*)d_in[9];
  const float* gru_bhh_f = (const float*)d_in[10];
  const float* gru_wih_b = (const float*)d_in[11];
  const float* gru_whh_b = (const float*)d_in[12];
  const float* gru_bih_b = (const float*)d_in[13];
  const float* gru_bhh_b = (const float*)d_in[14];
  const float* gate_fw   = (const float*)d_in[15];
  const float* gate_fb   = (const float*)d_in[16];
  const float* gate_bw   = (const float*)d_in[17];
  const float* ln_gru_w  = (const float*)d_in[18];
  const float* ln_gru_b  = (const float*)d_in[19];
  const float* moe_gw    = (const float*)d_in[20];
  const float* moe_gb    = (const float*)d_in[21];
  const float* moe_w1    = (const float*)d_in[22];
  const float* moe_b1    = (const float*)d_in[23];
  const float* moe_w2    = (const float*)d_in[24];
  const float* moe_b2    = (const float*)d_in[25];
  const float* ln_moe_w  = (const float*)d_in[26];
  const float* ln_moe_b  = (const float*)d_in[27];
  const float* mha_wi    = (const float*)d_in[28];
  const float* mha_bi    = (const float*)d_in[29];
  const float* mha_wo    = (const float*)d_in[30];
  const float* mha_bo    = (const float*)d_in[31];
  const float* ln_mha_w  = (const float*)d_in[32];
  const float* ln_mha_b  = (const float*)d_in[33];
  const float* gatp_w    = (const float*)d_in[34];
  const float* gatp_wu   = (const float*)d_in[35];
  const float* gatp_wv   = (const float*)d_in[36];
  const float* gatp_bias = (const float*)d_in[37];
  const float* gatp_pw   = (const float*)d_in[38];
  const float* gatp_pb   = (const float*)d_in[39];
  const float* gatn_w    = (const float*)d_in[40];
  const float* gatn_wu   = (const float*)d_in[41];
  const float* gatn_wv   = (const float*)d_in[42];
  const float* gatn_bias = (const float*)d_in[43];
  const float* gatn_pw   = (const float*)d_in[44];
  const float* gatn_pb   = (const float*)d_in[45];
  const float* hyp_tw    = (const float*)d_in[46];
  const float* hyp_tb    = (const float*)d_in[47];
  const float* hyp_pw    = (const float*)d_in[48];
  const float* hyp_lw    = (const float*)d_in[49];
  const float* hyp_lb    = (const float*)d_in[50];
  const float* sem_w1    = (const float*)d_in[51];
  const float* sem_b1    = (const float*)d_in[52];
  const float* sem_w2    = (const float*)d_in[53];
  const float* pred_w    = (const float*)d_in[54];
  const float* pred_b    = (const float*)d_in[55];

  constexpr size_t MB = 1u << 20;
  uint8_t* ws = (uint8_t*)d_ws;
  // big arenas (with lifetime-based reuse); peak requirement ~92 MB
  float*    x32   = (float*)(ws + 0);          // [0,16)
  _Float16* x16   = (_Float16*)(ws + 16 * MB); // [16,24)
  _Float16* zf16  = (_Float16*)(ws + 24 * MB); // [24,32)
  _Float16* zb16  = (_Float16*)(ws + 32 * MB); // [32,40)
  float*    gateS = (float*)(ws + 40 * MB);    // [40,56)
  float*    zg32  = (float*)(ws + 56 * MB);    // [56,72)
  _Float16* zg16  = (_Float16*)(ws + 72 * MB); // [72,80)
  // reuse (strictly after earlier producers are dead):
  float*    oute32 = (float*)(ws + 0);           // MoE temp, reuses x32
  _Float16* h1_16  = (_Float16*)(ws + 16 * MB);  // MoE hidden, reuses x16
  _Float16* qkv16  = (_Float16*)(ws + 0);        // 24MB, reuses x32+x16 (post MoE)
  float*    zm32   = (float*)(ws + 24 * MB);     // 16MB, reuses zf16+zb16
  _Float16* zm16   = (_Float16*)(ws + 40 * MB);  // 8MB, reuses gateS lo
  _Float16* za16   = (_Float16*)(ws + 48 * MB);  // 8MB, reuses gateS hi
  _Float16* hwht16 = (_Float16*)(ws + 72 * MB);  // 8MB, reuses zg16 (post MoE)

  uint8_t* SP = ws + 80 * MB;
  size_t so = 0;
  auto sal = [&](size_t bytes) -> void* {
    void* p = SP + so;
    so = (so + bytes + 255) & ~(size_t)255;
    return p;
  };
  // transposed (N x K) f16 weights
  _Float16* wihF = (_Float16*)sal(12288 * 2);
  _Float16* whhF = (_Float16*)sal(12288 * 2);
  _Float16* wihB = (_Float16*)sal(12288 * 2);
  _Float16* whhB = (_Float16*)sal(12288 * 2);
  _Float16* gfw  = (_Float16*)sal(4096 * 2);
  _Float16* gbw  = (_Float16*)sal(4096 * 2);
  _Float16* mw1  = (_Float16*)sal(16384 * 2);
  _Float16* mw2  = (_Float16*)sal(16384 * 2);
  _Float16* mwi  = (_Float16*)sal(12288 * 2);
  _Float16* mwo  = (_Float16*)sal(4096 * 2);
  _Float16* gpW  = (_Float16*)sal(4096 * 2);
  _Float16* gpPW = (_Float16*)sal(4096 * 2);
  _Float16* gnW  = (_Float16*)sal(4096 * 2);
  _Float16* gnPW = (_Float16*)sal(4096 * 2);
  _Float16* htw16 = (_Float16*)sal(2048 * 2);
  _Float16* hpw16 = (_Float16*)sal(4096 * 2);
  _Float16* sw116 = (_Float16*)sal(8192 * 2);
  int*   assign = (int*)sal(65536 * 4);
  float* wsel   = (float*)sal(65536 * 4);
  float* suP = (float*)sal(16384 * 4);
  float* svP = (float*)sal(16384 * 4);
  float* suN = (float*)sal(16384 * 4);
  float* svN = (float*)sal(16384 * 4);
  float* t1   = (float*)sal(65536 * 4);
  float* Hs32 = (float*)sal(65536 * 4);
  _Float16* Hs16  = (_Float16*)sal(65536 * 2);
  _Float16* HsW16 = (_Float16*)sal(65536 * 2);
  float* jsd  = (float*)sal(1024 * 4);
  float* wvec = (float*)sal(32 * 4);
  _Float16* hp16T = (_Float16*)sal(131072 * 2); // hp stored transposed (64 x 2048)
  float* logits = (float*)sal(8192 * 4);
  float* colmu  = (float*)sal(64 * 4);
  float* hproj  = (float*)sal(131072 * 4);
  float* zhyp   = (float*)sal(131072 * 4);
  float* h32    = (float*)sal(131072 * 4);
  _Float16* h16 = (_Float16*)sal(131072 * 2);
  float* outp   = (float*)sal(131072 * 4);
  float* outn   = (float*)sal(131072 * 4);
  float* hyp32  = (float*)sal(131072 * 4);
  float* sp32   = (float*)sal(131072 * 4);
  float* sstack32 = (float*)sal(524288 * 4);
  _Float16* sstack16 = (_Float16*)sal(524288 * 2);
  _Float16* t2_16 = (_Float16*)sal(1048576 * 2);
  float* fused = (float*)sal(131072 * 4);

  auto gemm = [&](const _Float16* A, int a_ld, const _Float16* Bt, int b_ld,
                  const float* bias, const float* bias2, float* C, int c_ld,
                  _Float16* C16, int c16_ld, int c16T,
                  int M, int Nc, int K, int act, int accum) {
    int ntiles = (M / 16) * (Nc / 16);
    k_gemm<<<(ntiles + 3) / 4, 128, 0, stream>>>(A, a_ld, Bt, b_ld, bias, bias2,
                                                 C, c_ld, C16, c16_ld, c16T,
                                                 M, Nc, K, act, accum);
  };
  auto f2hT = [&](const float* s, _Float16* d, int K, int N) {
    k_f2hT<<<(K * N + 255) / 256, 256, 0, stream>>>(s, d, K, N);
  };

  // ---- weight conversions to transposed f16 (N x K) ----
  f2hT(gru_wih_f, wihF, 64, 192); f2hT(gru_whh_f, whhF, 64, 192);
  f2hT(gru_wih_b, wihB, 64, 192); f2hT(gru_whh_b, whhB, 64, 192);
  f2hT(gate_fw, gfw, 64, 64);     f2hT(gate_bw, gbw, 64, 64);
  for (int e = 0; e < 4; ++e) {
    f2hT(moe_w1 + e * 4096, mw1 + e * 4096, 64, 64);
    f2hT(moe_w2 + e * 4096, mw2 + e * 4096, 64, 64);
  }
  f2hT(mha_wi, mwi, 64, 192);     f2hT(mha_wo, mwo, 64, 64);
  f2hT(gatp_w, gpW, 64, 64);      f2hT(gatp_pw, gpPW, 64, 64);
  f2hT(gatn_w, gnW, 64, 64);      f2hT(gatn_pw, gnPW, 64, 64);
  f2hT(hyp_tw, htw16, 64, 32);    f2hT(hyp_pw, hpw16, 64, 64);
  f2hT(sem_w1, sw116, 64, 128);

  // ---- input LN + projection ----
  k_inproj<<<256, 256, 0, stream>>>(features, ln_in_w, ln_in_b, proj_w, proj_b, x32, x16);

  // ---- bi-GRU (fused persistent recurrence, WMMA per step) ----
  k_gru<<<32, 128, 0, stream>>>(x16, wihF, whhF, gru_bih_f, gru_bhh_f, zf16, 0);
  k_gru<<<32, 128, 0, stream>>>(x16, wihB, whhB, gru_bih_b, gru_bhh_b, zb16, 1);

  // ---- gate = zf@fw + fb + zb@bw, then combine + LN ----
  gemm(zf16, 64, gfw, 64, gate_fb, nullptr, gateS, 64, nullptr, 0, 0, 65536, 64, 64, 0, 0);
  gemm(zb16, 64, gbw, 64, nullptr, nullptr, gateS, 64, nullptr, 0, 0, 65536, 64, 64, 0, 1);
  k_zg<<<8192, 256, 0, stream>>>(gateS, zf16, zb16, x32, ln_gru_w, ln_gru_b, zg32, zg16);

  // ---- MoE ----
  k_moe_gate<<<256, 256, 0, stream>>>(zg32, moe_gw, moe_gb, assign, wsel);
  for (int e = 0; e < 4; ++e) {
    gemm(zg16, 64, mw1 + e * 4096, 64, moe_b1 + e * 64, nullptr,
         nullptr, 0, h1_16, 64, 0, 65536, 64, 64, 1, 0); // gelu
    gemm(h1_16, 64, mw2 + e * 4096, 64, moe_b2 + e * 64, nullptr,
         oute32, 64, nullptr, 0, 0, 65536, 64, 64, 0, 0);
    k_moe_accum<<<16384, 256, 0, stream>>>(oute32, assign, wsel, zm32, e);
  }
  k_rowln<<<8192, 256, 0, stream>>>(zm32, 64, zg32, 64, ln_moe_w, ln_moe_b, zm32, zm16, 65536);

  // ---- MHA ----
  gemm(zm16, 64, mwi, 64, mha_bi, nullptr, nullptr, 0, qkv16, 192, 0, 65536, 192, 64, 0, 0);
  k_mha<<<1024, 128, 0, stream>>>(qkv16, za16);
  // out-projection only at t=31 (only last timestep feeds the graph stage)
  gemm(za16 + 31 * 64, 2048, mwo, 64, mha_bo, nullptr, hproj, 64, nullptr, 0, 0, 2048, 64, 64, 0, 0);
  k_rowln<<<256, 256, 0, stream>>>(hproj, 64, zm32 + 31 * 64, 2048, ln_mha_w, ln_mha_b, h32, h16, 2048);

  // ---- GAT (pos) ----
  gemm(h16, 64, gpW, 64, nullptr, nullptr, sp32, 64, nullptr, 0, 0, 2048, 64, 64, 0, 0);
  k_gat_uv<<<8, 256, 0, stream>>>(sp32, gatp_wu, gatp_wv, suP, svP);
  gemm(h16, 64, gpPW, 64, gatp_pb, gatp_bias, outp, 64, nullptr, 0, 0, 2048, 64, 64, 0, 0);
  k_gat<<<2048, 256, 0, stream>>>(pos_adj, sp32, suP, svP, outp);
  // ---- GAT (neg) ----
  gemm(h16, 64, gnW, 64, nullptr, nullptr, sp32, 64, nullptr, 0, 0, 2048, 64, 64, 0, 0);
  k_gat_uv<<<8, 256, 0, stream>>>(sp32, gatn_wu, gatn_wv, suN, svN);
  gemm(h16, 64, gnPW, 64, gatn_pb, gatn_bias, outn, 64, nullptr, 0, 0, 2048, 64, 64, 0, 0);
  k_gat<<<2048, 256, 0, stream>>>(neg_adj, sp32, suN, svN, outn);

  // ---- hypergraph ----
  gemm(h16, 64, htw16, 64, hyp_tb, nullptr, t1, 32, nullptr, 0, 0, 2048, 32, 64, 2, 0); // retanh
  k_colsoftmax<<<32, 256, 0, stream>>>(t1, Hs32, Hs16);
  k_jsd<<<128, 256, 0, stream>>>(Hs32, jsd);
  k_jsdw<<<1, 32, 0, stream>>>(jsd, wvec);
  k_hsw<<<256, 256, 0, stream>>>(Hs32, wvec, HsW16);
  // HWHT = (Hs*w) @ Hs^T : B^T storage is just Hs itself (2048 x 32)
  gemm(HsW16, 32, Hs16, 32, nullptr, nullptr, nullptr, 0, hwht16, 2048, 0, 2048, 2048, 32, 0, 0);
  // hp = h @ pw, stored transposed (64 x 2048) for the K=2048 GEMM below
  gemm(h16, 64, hpw16, 64, nullptr, nullptr, nullptr, 0, hp16T, 2048, 1, 2048, 64, 64, 0, 0);
  gemm(hwht16, 2048, hp16T, 2048, nullptr, nullptr, zhyp, 64, nullptr, 0, 0, 2048, 64, 2048, 0, 0);
  k_elu<<<512, 256, 0, stream>>>(zhyp);
  k_rowln<<<256, 256, 0, stream>>>(zhyp, 64, h32, 64, hyp_lw, hyp_lb, hyp32, nullptr, 2048);

  // ---- semantic fusion + prediction ----
  k_pack<<<512, 256, 0, stream>>>(h32, outp, outn, hyp32, sstack32, sstack16);
  gemm(sstack16, 64, sw116, 64, sem_b1, nullptr, nullptr, 0, t2_16, 128, 0, 8192, 128, 64, 3, 0); // tanh
  k_semlog<<<1024, 256, 0, stream>>>(t2_16, sem_w2, logits);
  k_fuse<<<256, 256, 0, stream>>>(logits, sstack32, fused);
  k_colmean<<<64, 256, 0, stream>>>(fused, colmu);
  k_final<<<256, 256, 0, stream>>>(fused, colmu, pred_w, pred_b, (float*)d_out);
}
